// MHC_18442589569996
// MI455X (gfx1250) — matrix operand order repaired and checked by reading, not compiled, not run on hardware
//
#include <hip/hip_runtime.h>
#include <stdint.h>

// ---------------- problem constants (from reference) ----------------
#define NSTREAMS   4
#define CDIM       2048
#define DDIM       8192            // n*C
#define NTOK_TILE  16              // tokens per workgroup (WMMA M)
#define NPAD       32              // padded projection columns (24 real)
#define KSTEP      32              // WMMA K per instruction (bf16)
#define EPSV       1e-5f
#define TMAXI      20

// LDS layout (bytes). lA rows padded by 8 halves (16B) to avoid 16KB-stride
// bank conflicts on the per-lane A-fragment ds_load_b128s.
#define AROW_HALVES (DDIM + 8)                         // 8200 halves / row
#define LA_BYTES    (NTOK_TILE * AROW_HALVES * 2)      // 262,400
#define LACC_OFF    (LA_BYTES)                         // 16x32 f32 = 2048
#define LSUM_OFF    (LACC_OFF + NTOK_TILE * NPAD * 4)  // 16 f32
#define LMIX_OFF    (LSUM_OFF + NTOK_TILE * 4)         // 16x16 f32 = 1024
#define SMEM_BYTES  (LMIX_OFF + NTOK_TILE * 16 * 4)    // ~265.5 KB

typedef __attribute__((ext_vector_type(16))) __bf16 v16bf;
typedef __attribute__((ext_vector_type(8)))  float  v8f;

struct U32x8 { uint4 lo, hi; };   // 32 bytes == one v16bf operand

static __device__ __forceinline__ unsigned short f2bf(float f) {
  uint32_t u = __float_as_uint(f);
  u += 0x7FFFu + ((u >> 16) & 1u);      // round-to-nearest-even
  return (unsigned short)(u >> 16);
}

// --------------------------------------------------------------------
// Prep: fold norm_w into [w_pre;w_post;w_res;zeros] (32 rows x 8192),
// bf16, stored directly in the B-operand lane layout:
//   bop[ ((kb*2+nt)*32 + lane)*16 + h ]  with  K = kb*32 + (lane>=16?16:0) + h,
//   N(row) = nt*16 + (lane&15).
// So phase-2 B fragment = two contiguous b128 loads per lane.
// --------------------------------------------------------------------
__global__ void mhc_prep(const float* __restrict__ norm_w,
                         const float* __restrict__ w_pre,
                         const float* __restrict__ w_post,
                         const float* __restrict__ w_res,
                         unsigned short* __restrict__ bop) {
  int tid  = blockIdx.x * blockDim.x + threadIdx.x;   // 262,144 total
  int h    = tid & 15;
  int lane = (tid >> 4) & 31;
  int nt   = (tid >> 9) & 1;
  int kb   = tid >> 10;
  int k    = kb * KSTEP + ((lane & 16) ? 16 : 0) + h;
  int row  = nt * 16 + (lane & 15);
  float w = 0.f;
  if      (row < 4)  w = w_pre [(row    ) * DDIM + k];
  else if (row < 8)  w = w_post[(row - 4) * DDIM + k];
  else if (row < 24) w = w_res [(row - 8) * DDIM + k];
  bop[tid] = f2bf(w * norm_w[k]);
}

// --------------------------------------------------------------------
// Fused: rmsnorm-folded projections (WMMA) + sigmoid/Sinkhorn + 4x4 mix.
// One workgroup = 16 tokens. 8 waves.
// --------------------------------------------------------------------
__global__ void __launch_bounds__(256, 1) mhc_fused(
    const float* __restrict__ x,
    const unsigned short* __restrict__ bop,
    const float* __restrict__ alpha_pre_p,
    const float* __restrict__ alpha_post_p,
    const float* __restrict__ alpha_res_p,
    const float* __restrict__ b_pre,
    const float* __restrict__ b_post,
    const float* __restrict__ b_res,
    float* __restrict__ out) {
  extern __shared__ char smem[];
  unsigned short* lA   = (unsigned short*)smem;          // [16][8200] bf16
  float*          lAcc = (float*)(smem + LACC_OFF);      // [16][32] raw dots
  float*          lSum = (float*)(smem + LSUM_OFF);      // [16] sum of squares
  float*          lMix = (float*)(smem + LMIX_OFF);      // [16][16] mix mat

  const int tid   = threadIdx.x;
  const int lane  = tid & 31;
  const int wave  = tid >> 5;              // 0..7
  const int tile0 = blockIdx.x * NTOK_TILE;

  // ---- phase 0: zero the cross-wave accumulation grid
  for (int i = tid; i < NTOK_TILE * NPAD; i += 256) lAcc[i] = 0.f;

  // ---- phase 1: stream x tile (f32), sum-of-squares + bf16 into LDS
  for (int tl = wave * 2; tl < wave * 2 + 2; ++tl) {
    const float4* xr = (const float4*)(x + (size_t)(tile0 + tl) * DDIM);
    unsigned short* dst = lA + tl * AROW_HALVES;
    float ss = 0.f;
    for (int i = lane; i < DDIM / 4; i += 32) {      // 512B/wave/iter, coalesced
      float4 v = xr[i];
      ss += v.x * v.x + v.y * v.y + v.z * v.z + v.w * v.w;
      uint2 p;
      p.x = (uint32_t)f2bf(v.x) | ((uint32_t)f2bf(v.y) << 16);
      p.y = (uint32_t)f2bf(v.z) | ((uint32_t)f2bf(v.w) << 16);
      *(uint2*)(dst + i * 4) = p;
    }
#pragma unroll
    for (int o = 16; o > 0; o >>= 1) ss += __shfl_xor(ss, o, 32);
    if (lane == 0) lSum[tl] = ss;
  }
  __syncthreads();

  // ---- phase 2: WMMA  [16 tok x 8192] x [8192 x 32]  (K split across waves)
  {
    const int mrow = lane & 15;
    const int ksel = (lane >> 4) & 1;   // lanes>=16 hold K offsets 16..31
    const unsigned short* aBase = lA + mrow * AROW_HALVES;
    v8f acc0 = {0.f, 0.f, 0.f, 0.f, 0.f, 0.f, 0.f, 0.f};
    v8f acc1 = {0.f, 0.f, 0.f, 0.f, 0.f, 0.f, 0.f, 0.f};
    for (int it = 0; it < DDIM / (8 * KSTEP); ++it) {   // 32 iterations
      const int k0 = it * 256 + wave * KSTEP;
      // A fragment (16-bit A 16x32 layout): two b128 from LDS
      U32x8 au;
      au.lo = *(const uint4*)(aBase + k0 + ksel * 16);
      au.hi = *(const uint4*)(aBase + k0 + ksel * 16 + 8);
      v16bf a = __builtin_bit_cast(v16bf, au);
      // B fragments: contiguous 32B per lane from pre-swizzled weights
      const int kb = k0 >> 5;
      const uint4* bp = (const uint4*)bop + ((size_t)kb * 64 + lane * 2);
      U32x8 b0u; b0u.lo = bp[0];  b0u.hi = bp[1];    // N cols 0..15
      U32x8 b1u; b1u.lo = bp[64]; b1u.hi = bp[65];   // N cols 16..31
      v16bf b0 = __builtin_bit_cast(v16bf, b0u);
      v16bf b1 = __builtin_bit_cast(v16bf, b1u);
      acc0 = __builtin_amdgcn_wmma_f32_16x16x32_bf16(false, a, false, b0,
                                                     (short)0, acc0, false, false);
      acc1 = __builtin_amdgcn_wmma_f32_16x16x32_bf16(false, a, false, b1,
                                                     (short)0, acc1, false, false);
    }
    // C/D layout: vgpr r, lanes<16 -> (M=r, N=lane); lanes>=16 -> (M=r+8, N=lane-16)
    const int mbase = (lane >> 4) ? 8 : 0;
    const int ncol  = lane & 15;
#pragma unroll
    for (int r = 0; r < 8; ++r) {
      atomicAdd(&lAcc[(mbase + r) * NPAD + ncol],      acc0[r]);   // ds_add_f32
      atomicAdd(&lAcc[(mbase + r) * NPAD + 16 + ncol], acc1[r]);
    }
  }
  __syncthreads();

  // ---- phase 3: per-token epilogue (16 lanes, one token each)
  if (tid < NTOK_TILE) {
    const int t = tid;
    const float s = rsqrtf(lSum[t] * (1.f / (float)DDIM) + EPSV); // rms scale
    const float a_pre = alpha_pre_p[0], a_post = alpha_post_p[0],
                a_res = alpha_res_p[0];
    float hpre[4], hpost[4];
#pragma unroll
    for (int j = 0; j < 4; ++j) {
      hpre[j]  = 1.f / (1.f + __expf(-(a_pre  * s * lAcc[t * NPAD + j]     + b_pre[j])));
      hpost[j] = 2.f / (1.f + __expf(-(a_post * s * lAcc[t * NPAD + 4 + j] + b_post[j])));
    }
    float M[16];
    float mx = -3.0e38f;
#pragma unroll
    for (int q = 0; q < 16; ++q) {
      M[q] = a_res * s * lAcc[t * NPAD + 8 + q] + b_res[q];
      mx = fmaxf(mx, M[q]);
    }
#pragma unroll
    for (int q = 0; q < 16; ++q) M[q] = __expf(M[q] - mx);
    for (int itr = 0; itr < TMAXI; ++itr) {            // Sinkhorn-Knopp, 4x4
#pragma unroll
      for (int i = 0; i < 4; ++i) {
        float rr = 1.f / (M[i*4] + M[i*4+1] + M[i*4+2] + M[i*4+3]);
        M[i*4] *= rr; M[i*4+1] *= rr; M[i*4+2] *= rr; M[i*4+3] *= rr;
      }
#pragma unroll
      for (int j = 0; j < 4; ++j) {
        float cr = 1.f / (M[j] + M[4+j] + M[8+j] + M[12+j]);
        M[j] *= cr; M[4+j] *= cr; M[8+j] *= cr; M[12+j] *= cr;
      }
    }
    // fused mix: out[i,c] = sum_j (res[i,j] + post[i]*pre[j]) * x[j,c]
#pragma unroll
    for (int i = 0; i < 4; ++i)
#pragma unroll
      for (int j = 0; j < 4; ++j)
        lMix[t * 16 + i * 4 + j] = M[i * 4 + j] + hpost[i] * hpre[j];
  }
  __syncthreads();

  // ---- phase 4: apply 4x4 mix over C (re-read f32 x; tile is L2-hot)
  for (int tl = wave * 2; tl < wave * 2 + 2; ++tl) {
    float mm[16];
#pragma unroll
    for (int q = 0; q < 16; ++q) mm[q] = lMix[tl * 16 + q];
    const float4* xr   = (const float4*)(x   + (size_t)(tile0 + tl) * DDIM);
    float4*       orow = (float4*)      (out + (size_t)(tile0 + tl) * DDIM);
    for (int ci = lane; ci < CDIM / 4; ci += 32) {
      float4 xs[4];
#pragma unroll
      for (int j = 0; j < 4; ++j) xs[j] = xr[j * (CDIM / 4) + ci];
#pragma unroll
      for (int i = 0; i < 4; ++i) {
        float4 o;
        o.x = mm[i*4]*xs[0].x + mm[i*4+1]*xs[1].x + mm[i*4+2]*xs[2].x + mm[i*4+3]*xs[3].x;
        o.y = mm[i*4]*xs[0].y + mm[i*4+1]*xs[1].y + mm[i*4+2]*xs[2].y + mm[i*4+3]*xs[3].y;
        o.z = mm[i*4]*xs[0].z + mm[i*4+1]*xs[1].z + mm[i*4+2]*xs[2].z + mm[i*4+3]*xs[3].z;
        o.w = mm[i*4]*xs[0].w + mm[i*4+1]*xs[1].w + mm[i*4+2]*xs[2].w + mm[i*4+3]*xs[3].w;
        orow[i * (CDIM / 4) + ci] = o;
      }
    }
  }
}

// --------------------------------------------------------------------
extern "C" void kernel_launch(void* const* d_in, const int* in_sizes, int n_in,
                              void* d_out, int out_size, void* d_ws, size_t ws_size,
                              hipStream_t stream) {
  const float* x      = (const float*)d_in[0];
  const float* norm_w = (const float*)d_in[1];
  const float* w_pre  = (const float*)d_in[2];
  const float* w_post = (const float*)d_in[3];
  const float* w_res  = (const float*)d_in[4];
  const float* a_pre  = (const float*)d_in[5];
  const float* a_post = (const float*)d_in[6];
  const float* a_res  = (const float*)d_in[7];
  const float* b_pre  = (const float*)d_in[8];
  const float* b_post = (const float*)d_in[9];
  const float* b_res  = (const float*)d_in[10];
  float* outp = (float*)d_out;
  unsigned short* bop = (unsigned short*)d_ws;   // 512 KB pre-swizzled bf16 weights

  // 1) fold norm_w into weights, bf16, B-operand swizzle (262,144 elems)
  mhc_prep<<<1024, 256, 0, stream>>>(norm_w, w_pre, w_post, w_res, bop);

  // 2) fused kernel: 16 tokens per block
  const int ntok    = out_size / DDIM;         // 8192
  const int nblocks = ntok / NTOK_TILE;        // 512
  (void)hipFuncSetAttribute(reinterpret_cast<const void*>(mhc_fused),
                            hipFuncAttributeMaxDynamicSharedMemorySize,
                            (int)SMEM_BYTES);
  mhc_fused<<<nblocks, 256, SMEM_BYTES, stream>>>(
      x, bop, a_pre, a_post, a_res, b_pre, b_post, b_res, outp);
}
